// GCN_17008070492360
// MI455X (gfx1250) — matrix operand order repaired and checked by reading, not compiled, not run on hardware
//
#include <hip/hip_runtime.h>

// ---------------------------------------------------------------------------
// GCN fused kernel for MI455X (gfx1250, wave32, WMMA).
// out = relu( (sum_k relu(l2n(x)@W1^T + l2n(nb_k)@W1^T)) @ W2^T ) @ Wc^T
// ---------------------------------------------------------------------------

typedef __attribute__((ext_vector_type(16))) _Float16 v16h;
typedef __attribute__((ext_vector_type(8)))  _Float16 v8h;
typedef __attribute__((ext_vector_type(8)))  float    v8f;

namespace {
constexpr int F   = 128;   // feature dim
constexpr int KN  = 16;    // neighbors per node
constexpr int H1  = 64;
constexpr int H2  = 32;
constexpr int C   = 10;
constexpr int WAVES = 4;          // waves per block
constexpr int BLOCK = WAVES * 32; // wave32
// bank-conflict-avoiding LDS row strides (halves); all multiples of 8 (16B align)
constexpr int FP  = 136;   // padded F
constexpr int H1P = 72;    // padded H1
constexpr int H2P = 40;    // padded H2
}

__device__ __forceinline__ v8f wmma_f16(v16h a, v16h b, v8f c) {
  // D = A(16x32 f16) * B(32x16 f16) + C(16x16 f32)
  return __builtin_amdgcn_wmma_f32_16x16x32_f16(
      /*neg_a=*/false, a, /*neg_b=*/false, b,
      /*c_mod=*/(short)0, c, /*reuse_a=*/false, /*reuse_b=*/false);
}

// A-matrix fragment (16xK row-striped): lane l<16 holds row l, K {0..7,16..23}
// of the chunk; lane l+16 holds row l, K {8..15,24..31}.
__device__ __forceinline__ v16h load_afrag(const _Float16* p0, int rowStride,
                                           int kchunk, int lane) {
  const int row = lane & 15;
  const int k0  = kchunk * 32 + ((lane >> 4) << 3);
  const _Float16* p = p0 + row * rowStride + k0;
  union U { v16h v; v8h h[2]; } u;
  u.h[0] = *(const v8h*)(p);
  u.h[1] = *(const v8h*)(p + 16);
  return u.v;
}

// B-matrix fragment (KxN, column-per-lane): lane l<16 holds column l K {0..7,16..23}
// of the chunk, lane l+16 holds column l K {8..15,24..31}. B column n == W row n
// (since B = W^T), so gathers are contiguous 16B runs of the W row.
__device__ __forceinline__ v16h load_bfrag(const _Float16* w, int rowStride,
                                           int ntile, int kchunk, int lane) {
  const int col = lane & 15;
  const int k0  = kchunk * 32 + ((lane >> 4) << 3);
  const _Float16* p = w + (ntile * 16 + col) * rowStride + k0;
  union U { v16h v; v8h h[2]; } u;
  u.h[0] = *(const v8h*)(p);
  u.h[1] = *(const v8h*)(p + 16);
  return u.v;
}

// Load 16 rows of 128 f32, L2-normalize each row, store f16 into LDS tile
// (row stride FP). Two rows per pass: lanes 0-15 -> row 2i, 16-31 -> row 2i+1,
// 8 contiguous floats per lane (2x b128 global loads, fully coalesced).
__device__ __forceinline__ void load_normalize_rows(const float* __restrict__ src,
                                                    _Float16* dst, int lane) {
  const int li  = lane & 15;
  const int sub = lane >> 4;
#pragma unroll
  for (int it = 0; it < 8; ++it) {
    const int r = it * 2 + sub;
    const float* p = src + r * F + li * 8;
    float4 a = *(const float4*)(p);
    float4 b = *(const float4*)(p + 4);
    float ss = a.x*a.x + a.y*a.y + a.z*a.z + a.w*a.w
             + b.x*b.x + b.y*b.y + b.z*b.z + b.w*b.w;
    ss += __shfl_xor(ss, 1, 32);
    ss += __shfl_xor(ss, 2, 32);
    ss += __shfl_xor(ss, 4, 32);
    ss += __shfl_xor(ss, 8, 32);
    const float inv = 1.0f / fmaxf(sqrtf(ss), 1e-12f);
    v8h h;
    h[0] = (_Float16)(a.x * inv); h[1] = (_Float16)(a.y * inv);
    h[2] = (_Float16)(a.z * inv); h[3] = (_Float16)(a.w * inv);
    h[4] = (_Float16)(b.x * inv); h[5] = (_Float16)(b.y * inv);
    h[6] = (_Float16)(b.z * inv); h[7] = (_Float16)(b.w * inv);
    *(v8h*)(dst + r * FP + li * 8) = h;
  }
}

__global__ __launch_bounds__(BLOCK) void gcn_fused_kernel(
    const float* __restrict__ x, const float* __restrict__ nb,
    const float* __restrict__ W1, const float* __restrict__ W2,
    const float* __restrict__ Wc, float* __restrict__ out, int nGroups) {
  // block-shared f16 weights (padded rows for bank spread)
  __shared__ _Float16 sW1[H1 * FP];        // 17408 B
  __shared__ _Float16 sW2[H2 * H1P];       //  4608 B
  __shared__ _Float16 sWc[16 * H2P];       //  1280 B (rows 10..15 zero)
  // per-wave private tiles (no barriers needed: DS in-order within wave)
  __shared__ _Float16 sA [WAVES][16 * FP];   // normalized row staging
  __shared__ _Float16 sX1[WAVES][16 * H1];   // x1 tile (f16)
  __shared__ _Float16 sS [WAVES][16 * H1P];  // aggregated S (f16)
  __shared__ _Float16 sH [WAVES][16 * H2P];  // relu(S@W2^T) (f16)

  const int tid = threadIdx.x;
  for (int i = tid; i < H1 * F; i += BLOCK) {
    sW1[(i >> 7) * FP + (i & 127)] = (_Float16)W1[i];
  }
  for (int i = tid; i < H2 * H1; i += BLOCK) {
    sW2[(i >> 6) * H1P + (i & 63)] = (_Float16)W2[i];
  }
  for (int i = tid; i < 16 * H2; i += BLOCK) {
    const int r = i >> 5, c2 = i & 31;
    sWc[r * H2P + c2] = (r < C) ? (_Float16)Wc[r * H2 + c2] : (_Float16)0.0f;
  }
  __syncthreads();

  const int wid  = tid >> 5;
  const int lane = tid & 31;
  _Float16* myA  = sA[wid];
  _Float16* myX1 = sX1[wid];
  _Float16* myS  = sS[wid];
  _Float16* myH  = sH[wid];

  const int col   = lane & 15;       // D-layout: column = lane&15
  const int rbase = (lane >> 4) * 8; // D-layout: lanes>=16 hold rows 8..15

  for (int g = blockIdx.x * WAVES + wid; g < nGroups;
       g += (int)gridDim.x * WAVES) {
    const int base = g * 16;

    // ---- x1 = l2norm(x[base..base+15]) @ W1^T  (16x64), cached as f16 ----
    load_normalize_rows(x + (size_t)base * F, myA, lane);
    {
      v8f accX[4] = {};
#pragma unroll
      for (int c = 0; c < 4; ++c) {
        v16h a = load_afrag(myA, FP, c, lane);
#pragma unroll
        for (int t = 0; t < 4; ++t)
          accX[t] = wmma_f16(a, load_bfrag(sW1, FP, t, c, lane), accX[t]);
      }
#pragma unroll
      for (int t = 0; t < 4; ++t)
#pragma unroll
        for (int v = 0; v < 8; ++v)
          myX1[(rbase + v) * H1 + t * 16 + col] = (_Float16)accX[t][v];
    }

    // ---- per node: nb1 tile, S[i] = sum_rows relu(x1[i] + nb1) ----
    const float* nbase = nb + (size_t)base * (KN * F);
    for (int i = 0; i < 16; ++i) {
      if (i + 1 < 16)  // pull next node's 8KB neighbor block toward L2/L0
        __builtin_prefetch(nbase + (size_t)(i + 1) * KN * F + lane * 64, 0, 0);
      load_normalize_rows(nbase + (size_t)i * KN * F, myA, lane);
      v8f accN[4] = {};
#pragma unroll
      for (int c = 0; c < 4; ++c) {
        v16h a = load_afrag(myA, FP, c, lane);
#pragma unroll
        for (int t = 0; t < 4; ++t)
          accN[t] = wmma_f16(a, load_bfrag(sW1, FP, t, c, lane), accN[t]);
      }
#pragma unroll
      for (int t = 0; t < 4; ++t) {
        const float bx = (float)myX1[i * H1 + t * 16 + col];  // x1[i, t*16+col]
        float part = 0.0f;
#pragma unroll
        for (int v = 0; v < 8; ++v) part += fmaxf(accN[t][v] + bx, 0.0f);
        part += __shfl_xor(part, 16, 32);  // rows 0-7 + rows 8-15
        if (lane < 16) myS[i * H1P + t * 16 + lane] = (_Float16)part;
      }
    }

    // ---- h = relu(S @ W2^T)  (16x32) ----
    {
      v8f acc2[2] = {};
#pragma unroll
      for (int c = 0; c < 2; ++c) {
        v16h a = load_afrag(myS, H1P, c, lane);
#pragma unroll
        for (int t = 0; t < 2; ++t)
          acc2[t] = wmma_f16(a, load_bfrag(sW2, H1P, t, c, lane), acc2[t]);
      }
#pragma unroll
      for (int t = 0; t < 2; ++t)
#pragma unroll
        for (int v = 0; v < 8; ++v)
          myH[(rbase + v) * H2P + t * 16 + col] =
              (_Float16)fmaxf(acc2[t][v], 0.0f);
    }

    // ---- out = h @ Wc^T  (16x10, N padded to 16 with zero cols) ----
    {
      v8f acc3 = {};
      acc3 = wmma_f16(load_afrag(myH, H2P, 0, lane),
                      load_bfrag(sWc, H2P, 0, 0, lane), acc3);
      if (col < C) {
#pragma unroll
        for (int v = 0; v < 8; ++v)
          out[(size_t)(base + rbase + v) * C + col] = acc3[v];
      }
    }
  }
}

extern "C" void kernel_launch(void* const* d_in, const int* in_sizes, int n_in,
                              void* d_out, int out_size, void* d_ws,
                              size_t ws_size, hipStream_t stream) {
  const float* x  = (const float*)d_in[0];  // [N, 128]
  const float* nb = (const float*)d_in[1];  // [N, 16, 128]
  const float* W1 = (const float*)d_in[2];  // [64, 128]
  const float* W2 = (const float*)d_in[3];  // [32, 64]
  const float* Wc = (const float*)d_in[4];  // [10, 32]
  float* out = (float*)d_out;               // [N, 10]

  const int N = in_sizes[0] / F;
  const int nGroups = N / 16;  // N = 100000 -> 6250 groups, exact
  if (nGroups <= 0) return;
  const int blocks = (nGroups + WAVES - 1) / WAVES;  // one group per wave
  gcn_fused_kernel<<<blocks, BLOCK, 0, stream>>>(x, nb, W1, W2, Wc, out,
                                                 nGroups);
}
